// PflowModel_84696755077243
// MI455X (gfx1250) — matrix operand order repaired and checked by reading, not compiled, not run on hardware
//
#include <hip/hip_runtime.h>

#define NODES 1048576
#define NSEG  2048
#define DD    128

typedef _Float16 half8_t __attribute__((ext_vector_type(8)));
typedef _Float16 v16h    __attribute__((ext_vector_type(16)));
typedef float    v8f     __attribute__((ext_vector_type(8)));

__device__ __forceinline__ float leaky1(float x) { return x > 0.f ? x : 0.01f * x; }

// ---------------------------------------------------------------------------
// net1 first layer: t = leaky(feat @ W1a + b1a); feat [N,3] f32 -> out [N,128] f16
// one thread per column, loop rows; W1a column cached in regs.
// ---------------------------------------------------------------------------
__global__ void k_feat(const float* __restrict__ feat, const float* __restrict__ W1a,
                       const float* __restrict__ b1a, _Float16* __restrict__ out,
                       int nrows, int rpb) {
    int t  = threadIdx.x;                 // column 0..127
    float w0 = W1a[t], w1 = W1a[DD + t], w2 = W1a[2 * DD + t], bb = b1a[t];
    int base = blockIdx.x * rpb;
    int end  = base + rpb; if (end > nrows) end = nrows;
    for (int r = base; r < end; ++r) {
        float f0 = feat[(size_t)r * 3 + 0];
        float f1 = feat[(size_t)r * 3 + 1];
        float f2 = feat[(size_t)r * 3 + 2];
        float x  = fmaf(f0, w0, fmaf(f1, w1, fmaf(f2, w2, bb)));
        out[(size_t)r * DD + t] = (_Float16)leaky1(x);
    }
}

// ---------------------------------------------------------------------------
// Transpose + f16-convert one [128,128] f32 weight: Wt[n][k] = W[k][n]
// ---------------------------------------------------------------------------
__global__ void k_prepw(const float* __restrict__ W, _Float16* __restrict__ Wt) {
    int k = blockIdx.x, n = threadIdx.x;
    Wt[n * DD + k] = (_Float16)W[k * DD + n];
}

// ---------------------------------------------------------------------------
// Segment counts via run-length + atomics (seg ids are sorted)
// ---------------------------------------------------------------------------
__global__ void k_counts(const int* __restrict__ seg, int* __restrict__ cnt, int n) {
    int i0 = (blockIdx.x * blockDim.x + threadIdx.x) * 64;
    if (i0 >= n) return;
    int iend = i0 + 64; if (iend > n) iend = n;
    int cur = seg[i0]; int c = 0;
    for (int i = i0; i < iend; ++i) {
        int s = seg[i];
        if (s != cur) { atomicAdd(&cnt[cur], c); c = 0; cur = s; }
        ++c;
    }
    atomicAdd(&cnt[cur], c);
}

__global__ void k_invcnt(const int* __restrict__ cnt, float* __restrict__ inv, int g) {
    int i = blockIdx.x * blockDim.x + threadIdx.x;
    if (i < g) inv[i] = 1.0f / fmaxf((float)cnt[i], 1.0f);
}

// ---------------------------------------------------------------------------
// Segment sums: block owns 256 contiguous rows, thread t owns column t.
// Run-length accumulate in registers, flush with one atomicAdd per run.
// ---------------------------------------------------------------------------
__global__ void k_segreduce(const _Float16* __restrict__ h, const int* __restrict__ seg,
                            float* __restrict__ S, int nrows, int rpb) {
    int t    = threadIdx.x;               // column 0..127
    int base = blockIdx.x * rpb;
    int end  = base + rpb; if (end > nrows) end = nrows;
    if (base >= end) return;
    int   cur = seg[base];
    float acc = 0.f;
    for (int r = base; r < end; ++r) {
        int sg = seg[r];
        if (sg != cur) { atomicAdd(&S[cur * DD + t], acc); acc = 0.f; cur = sg; }
        acc += (float)h[(size_t)r * DD + t];
    }
    atomicAdd(&S[cur * DD + t], acc);
}

// ---------------------------------------------------------------------------
// Fused layer: out = leaky((A (+ S[seg]/cnt)) @ W + b), D=128, WMMA f16.
// One wave per 16-row tile. In-place safe (out may alias A: rows independent,
// all A fragments loaded before any store).
// A frag (16x32 f16, ISA layout): lane row = lane&15; lanes>=16 take K+8;
// elems 0..7 -> K+0..7, elems 8..15 -> K+16..23 within a 32-wide chunk.
// B frag from Wt[n][k] (pre-transposed): same K layout as A (lanes>=16 K+8;
// elems 0..7 -> K+0..7, 8..15 -> K+16..23), col n = j*16 + (lane&15).
// C/D: col = lane&15, row M = vgpr + 8*(lane>>4).
// ---------------------------------------------------------------------------
__global__ void k_gemm_d128(const _Float16* __restrict__ A,
                            const float* __restrict__ S,        // null => no residual
                            const float* __restrict__ invcnt,
                            const int* __restrict__ seg,
                            const _Float16* __restrict__ Wt,    // [128 n][128 k] f16
                            const float* __restrict__ bias,
                            _Float16* __restrict__ out,
                            int nrows) {
    const int lane  = threadIdx.x & 31;
    const int wave  = blockIdx.x * (blockDim.x >> 5) + (threadIdx.x >> 5);
    const int r0    = wave * 16;
    if (r0 >= nrows) return;                  // wave-uniform; EXEC stays all-ones
    const int mrow  = r0 + (lane & 15);
    const int khalf = lane >> 4;              // 0: K+0/K+16 ; 1: K+8/K+24

    const _Float16* Arow = A + (size_t)mrow * DD;

    // Residual mean for this lane's row (optional)
    const float* Sr = nullptr;
    float inv = 0.f;
    if (S) {
        int sg = seg[mrow];
        Sr  = S + sg * DD;
        inv = invcnt[sg];
    }

    // Load all 4 A fragments up front (enables in-place out==A)
    v16h a[4];
#pragma unroll
    for (int kc = 0; kc < 4; ++kc) {
        int kb = kc * 32 + khalf * 8;
        half8_t lo = *(const half8_t*)(Arow + kb);
        half8_t hi = *(const half8_t*)(Arow + kb + 16);
        if (S) {
#pragma unroll
            for (int e = 0; e < 8; ++e) {
                lo[e] = (_Float16)((float)lo[e] + Sr[kb + e]      * inv);
                hi[e] = (_Float16)((float)hi[e] + Sr[kb + 16 + e] * inv);
            }
        }
        a[kc] = __builtin_shufflevector(lo, hi, 0, 1, 2, 3, 4, 5, 6, 7,
                                                8, 9, 10, 11, 12, 13, 14, 15);
    }

    const int ncol = lane & 15;
    for (int j = 0; j < 8; ++j) {             // 8 column tiles of 16
        const _Float16* Wn = Wt + (size_t)(j * 16 + ncol) * DD + khalf * 8;
        v8f c = {};
#pragma unroll
        for (int kc = 0; kc < 4; ++kc) {
            half8_t blo = *(const half8_t*)(Wn + kc * 32);
            half8_t bhi = *(const half8_t*)(Wn + kc * 32 + 16);
            v16h b = __builtin_shufflevector(blo, bhi, 0, 1, 2, 3, 4, 5, 6, 7,
                                                       8, 9, 10, 11, 12, 13, 14, 15);
            c = __builtin_amdgcn_wmma_f32_16x16x32_f16(false, a[kc], false, b,
                                                       (short)0, c, false, false);
        }
        float bv = bias[j * 16 + ncol];
        _Float16* op = out + (size_t)(r0 + 8 * khalf) * DD + j * 16 + ncol;
#pragma unroll
        for (int r = 0; r < 8; ++r) {
            float x = c[r] + bv;
            op[(size_t)r * DD] = (_Float16)leaky1(x);
        }
    }
}

// ---------------------------------------------------------------------------
// Output head: g = S5/cnt [2048,128]; energy = leaky(g@Wo1+bo1) @ Wo2 + bo2
// One block (64 threads) per graph; thread t owns hidden unit t.
// ---------------------------------------------------------------------------
__global__ void k_out(const float* __restrict__ S, const float* __restrict__ invcnt,
                      const float* __restrict__ Wo1, const float* __restrict__ bo1,
                      const float* __restrict__ Wo2, const float* __restrict__ bo2,
                      float* __restrict__ outp) {
    int g = blockIdx.x, t = threadIdx.x;      // t in 0..63
    float inv = invcnt[g];
    const float* Sr = S + g * DD;
    float acc = bo1[t];
    for (int k = 0; k < DD; ++k) acc = fmaf(Sr[k] * inv, Wo1[k * 64 + t], acc);
    float v = leaky1(acc) * Wo2[t];
    __shared__ float red[64];
    red[t] = v; __syncthreads();
    for (int s = 32; s > 0; s >>= 1) { if (t < s) red[t] += red[t + s]; __syncthreads(); }
    if (t == 0) outp[g] = red[0] + bo2[0];
}

// ---------------------------------------------------------------------------
extern "C" void kernel_launch(void* const* d_in, const int* in_sizes, int n_in,
                              void* d_out, int out_size, void* d_ws, size_t ws_size,
                              hipStream_t stream) {
    const float* feat = (const float*)d_in[0];
    const int*   seg  = (const int*)  d_in[1];
    const float* W1a  = (const float*)d_in[2];
    const float* b1a  = (const float*)d_in[3];
    const float* Wlin[5] = { (const float*)d_in[4], (const float*)d_in[6],
                             (const float*)d_in[8], (const float*)d_in[10],
                             (const float*)d_in[12] };            // W1b, W2..W5
    const float* blin[5] = { (const float*)d_in[5], (const float*)d_in[7],
                             (const float*)d_in[9], (const float*)d_in[11],
                             (const float*)d_in[13] };            // b1b, b2..b5
    const float* Wo1 = (const float*)d_in[14];
    const float* bo1 = (const float*)d_in[15];
    const float* Wo2 = (const float*)d_in[16];
    const float* bo2 = (const float*)d_in[17];

    // workspace carve-up (~258 MB)
    char* ws = (char*)d_ws;
    _Float16* h = (_Float16*)ws;                       size_t off = (size_t)NODES * DD * 2;
    float* S    = (float*)(ws + off);                  off += (size_t)NSEG * DD * 4;
    int*   cnt  = (int*)  (ws + off);                  off += NSEG * 4;
    float* invc = (float*)(ws + off);                  off += NSEG * 4;
    _Float16* Wt[5];
    for (int i = 0; i < 5; ++i) { Wt[i] = (_Float16*)(ws + off); off += (size_t)DD * DD * 2; }

    // --- one-time prep (cheap, re-run every call for determinism) ---
    hipMemsetAsync(cnt, 0, NSEG * 4, stream);
    for (int i = 0; i < 5; ++i)
        k_prepw<<<DD, DD, 0, stream>>>(Wlin[i], Wt[i]);
    {
        int nth = (NODES + 63) / 64;
        k_counts<<<(nth + 255) / 256, 256, 0, stream>>>(seg, cnt, NODES);
        k_invcnt<<<(NSEG + 255) / 256, 256, 0, stream>>>(cnt, invc, NSEG);
    }

    // --- net1: feat -> t -> h1 (in-place GEMM on h) ---
    k_feat<<<NODES / 8, DD, 0, stream>>>(feat, W1a, b1a, h, NODES, 8);
    const int gemm_blocks = (NODES / 16) / 8;          // 8 waves/block, 1 tile/wave
    k_gemm_d128<<<gemm_blocks, 256, 0, stream>>>(h, nullptr, nullptr, nullptr,
                                                 Wt[0], blin[0], h, NODES);

    // --- net2..net5: reduce -> fused (residual-mean + GEMM + bias + leaky) ---
    for (int l = 1; l < 5; ++l) {
        hipMemsetAsync(S, 0, (size_t)NSEG * DD * 4, stream);
        k_segreduce<<<NODES / 256, DD, 0, stream>>>(h, seg, S, NODES, 256);
        k_gemm_d128<<<gemm_blocks, 256, 0, stream>>>(h, S, invc, seg,
                                                     Wt[l], blin[l], h, NODES);
    }

    // --- final per-graph mean + output head ---
    hipMemsetAsync(S, 0, (size_t)NSEG * DD * 4, stream);
    k_segreduce<<<NODES / 256, DD, 0, stream>>>(h, seg, S, NODES, 256);
    k_out<<<NSEG, 64, 0, stream>>>(S, invc, Wo1, bo1, Wo2, bo2, (float*)d_out);
}
